// RnnEncoderBlock_70815420776725
// MI455X (gfx1250) — matrix operand order repaired and checked by reading, not compiled, not run on hardware
//
#include <hip/hip_runtime.h>
#include <math.h>

// ---------------------------------------------------------------------------
// RNN encoder for MI455X (gfx1250, wave32, WMMA).
//   Phase 1: xU = x @ U + b    (bandwidth-bound GEMM, f16 WMMA, f32 accum)
//   Phase 2: h_t = tanh(xU_t + h_{t-1} @ V)   (persistent kernel, grid barrier,
//            V slice resident in LDS, h carried as f16 in workspace)
// ---------------------------------------------------------------------------

typedef __attribute__((ext_vector_type(16))) _Float16 v16h;
typedef __attribute__((ext_vector_type(8)))  _Float16 v8h;
typedef __attribute__((ext_vector_type(8)))  float    v8f;

#define B_  64
#define T_  512
#define H_  512
#define M_TOTAL (B_ * T_)          // 32768

// workspace layout (bytes)
#define WS_UT   0                          // U^T as f16 [N=512][K=512]
#define WS_VT   (512 * 512 * 2)            // V^T as f16 [N=512][K=512]
#define WS_H    (WS_VT + 512 * 512 * 2)    // h double buffer, 2 x [64][512] f16
#define WS_CNT  (WS_H + 2 * B_ * H_ * 2)   // barrier counter
#define WS_NEED (WS_CNT + 64)

// ---- WMMA fragment loaders (wave32 layouts, CDNA5 ISA 7.12.2) --------------
// A (16-bit, 16x32): lane L holds row M=L%16; elems 0..7 = K kb+(L/16)*8..+7,
//                    elems 8..15 = K kb+16+(L/16)*8..+7.
__device__ __forceinline__ v16h ld_frag_a(const _Float16* row, int kb, int sub) {
    v8h lo = *(const v8h*)(row + kb + sub * 8);
    v8h hi = *(const v8h*)(row + kb + 16 + sub * 8);
    v16h r;
#pragma unroll
    for (int i = 0; i < 8; ++i) { r[i] = lo[i]; r[i + 8] = hi[i]; }
    return r;
}
// B (16-bit, 32x16): lane L holds column N=L%16; elems 0..7 = K kb+(L/16)*8..+7,
// elems 8..15 = K kb+16+(L/16)*8..+7
// (column-major source, i.e. V^T / U^T rows are contiguous in K).
__device__ __forceinline__ v16h ld_frag_b(const _Float16* coln, int kb, int sub) {
    v8h lo = *(const v8h*)(coln + kb + sub * 8);
    v8h hi = *(const v8h*)(coln + kb + 16 + sub * 8);
    v16h r;
#pragma unroll
    for (int i = 0; i < 8; ++i) { r[i] = lo[i]; r[i + 8] = hi[i]; }
    return r;
}

// ---- prep: transpose+convert weights to f16, convert h0, zero barrier ------
__global__ void prep_kernel(const float* __restrict__ U, const float* __restrict__ V,
                            const float* __restrict__ h0,
                            _Float16* __restrict__ Ut, _Float16* __restrict__ Vt,
                            _Float16* __restrict__ hb, unsigned* __restrict__ cnt) {
    int id = blockIdx.x * blockDim.x + threadIdx.x;
    if (id == 0) *cnt = 0u;
    const int stride = gridDim.x * blockDim.x;
    for (int i = id; i < 512 * 512; i += stride) {
        int k = i >> 9, c = i & 511;                 // coalesced read, strided write
        Ut[c * 512 + k] = (_Float16)U[i];
        Vt[c * 512 + k] = (_Float16)V[i];
    }
    for (int i = id; i < B_ * H_; i += stride)
        hb[i] = (_Float16)h0[i];                     // h buffer 0 = h0
}

// ---- phase 1: xU = x @ U + bias  (block 128x128, 8 waves, wave tile 64x32) --
__global__ __launch_bounds__(256) void gemm_xu_kernel(
    const float* __restrict__ x,        // [32768][512] f32
    const _Float16* __restrict__ Ut,    // [512][512]  f16, [N][K]
    const float* __restrict__ bias,     // [512]
    float* __restrict__ out)            // [32768][512] f32  (holds xU)
{
    __shared__ _Float16 sA[128][40];    // 40-elem stride: 16B-aligned, bank-spread
    __shared__ _Float16 sB[128][40];
    const int tid  = threadIdx.x;
    const int lane = tid & 31, wv = tid >> 5;
    const int sub  = lane >> 4, ln16 = lane & 15;
    const int wm = wv >> 2, wn = wv & 3;            // waves 2(M) x 4(N)
    const int bm = blockIdx.x & 255;                // 256 M-blocks
    const int bn = blockIdx.x >> 8;                 // 4 N-blocks
    const int rowBase = bm * 128, colBase = bn * 128;

    v8f acc[4][2];
#pragma unroll
    for (int mt = 0; mt < 4; ++mt)
#pragma unroll
        for (int nt = 0; nt < 2; ++nt) acc[mt][nt] = (v8f)0.0f;

    for (int kc = 0; kc < 16; ++kc) {
        __syncthreads();
        // stage A tile 128x32 (f32 -> f16)
#pragma unroll
        for (int it = 0; it < 4; ++it) {
            int idx = tid + it * 256;               // 1024 float4 chunks
            int r = idx >> 3, q = idx & 7;
            float4 v = *(const float4*)(x + (size_t)(rowBase + r) * 512 + kc * 32 + q * 4);
            _Float16* dp = &sA[r][q * 4];
            dp[0] = (_Float16)v.x; dp[1] = (_Float16)v.y;
            dp[2] = (_Float16)v.z; dp[3] = (_Float16)v.w;
        }
        // stage B tile: 128 N-rows x 32 K (already f16)
#pragma unroll
        for (int it = 0; it < 2; ++it) {
            int idx = tid + it * 256;               // 512 v8h chunks
            int c = idx >> 2, q = idx & 3;
            *(v8h*)&sB[c][q * 8] = *(const v8h*)(Ut + (size_t)(colBase + c) * 512 + kc * 32 + q * 8);
        }
        __syncthreads();

        v16h a[4], b[2];
#pragma unroll
        for (int mt = 0; mt < 4; ++mt) a[mt] = ld_frag_a(&sA[wm * 64 + mt * 16 + ln16][0], 0, sub);
#pragma unroll
        for (int nt = 0; nt < 2; ++nt) b[nt] = ld_frag_b(&sB[wn * 32 + nt * 16 + ln16][0], 0, sub);
#pragma unroll
        for (int mt = 0; mt < 4; ++mt)
#pragma unroll
            for (int nt = 0; nt < 2; ++nt)
                acc[mt][nt] = __builtin_amdgcn_wmma_f32_16x16x32_f16(
                    false, a[mt], false, b[nt], (short)0, acc[mt][nt], false, false);
    }
    // epilogue: + bias, f32 store (C layout: m = r + 8*sub, n = ln16)
#pragma unroll
    for (int nt = 0; nt < 2; ++nt) {
        const int col = colBase + wn * 32 + nt * 16 + ln16;
        const float bv = bias[col];
#pragma unroll
        for (int mt = 0; mt < 4; ++mt)
#pragma unroll
            for (int r = 0; r < 8; ++r) {
                const int row = rowBase + wm * 64 + mt * 16 + sub * 8 + r;
                out[(size_t)row * 512 + col] = acc[mt][nt][r] + bv;
            }
    }
}

// ---- phase 2: persistent recurrence, 16 WGs x 256 thr, grid barrier --------
__global__ __launch_bounds__(256) void rnn_persistent_kernel(
    const _Float16* __restrict__ Vt,    // [512][512] f16, [N][K]
    _Float16* __restrict__ hbuf,        // 2 x [64][512] f16
    float* __restrict__ out,            // [64][512][512] f32 (holds xU, gets h)
    unsigned* __restrict__ counter, unsigned nwg)
{
    __shared__ _Float16 sV[32][520];    // this WG's 32 columns of V^T, padded
    const int tid  = threadIdx.x;
    const int lane = tid & 31, wv = tid >> 5;       // 8 waves
    const int sub  = lane >> 4, ln16 = lane & 15;
    const int mt = wv & 3, nt = wv >> 2;            // tiles: 4(M) x 2(N)
    const int n0 = blockIdx.x * 32;

    // load V^T slice once (32 x 512 f16 = 32 KB -> stays in LDS for all 512 steps)
#pragma unroll
    for (int it = 0; it < 8; ++it) {
        int idx = tid + it * 256;                   // 2048 v8h chunks
        int c = idx >> 6, q = idx & 63;
        *(v8h*)&sV[c][q * 8] = *(const v8h*)(Vt + (size_t)(n0 + c) * 512 + q * 8);
    }
    __syncthreads();

    const int col = n0 + nt * 16 + ln16;
    const _Float16* bcol = &sV[nt * 16 + ln16][0];
    const int mrow = mt * 16;

    for (int t = 0; t < T_; ++t) {
        const _Float16* hin  = hbuf + (size_t)(t & 1) * (B_ * H_);
        _Float16*       hout = hbuf + (size_t)((t + 1) & 1) * (B_ * H_);

        // accumulator starts at xU[b, t, col] (already contains bias)
        v8f acc;
#pragma unroll
        for (int r = 0; r < 8; ++r) {
            const int m = mrow + sub * 8 + r;       // batch index
            acc[r] = out[(size_t)m * (T_ * H_) + (size_t)t * H_ + col];
        }
        // h_{t-1} @ V : K = 512 -> 16 chained WMMAs
        const _Float16* arow = hin + (size_t)(mrow + ln16) * H_;
#pragma unroll
        for (int kc = 0; kc < 16; ++kc) {
            v16h a = ld_frag_a(arow, kc * 32, sub);
            v16h b = ld_frag_b(bcol, kc * 32, sub);
            acc = __builtin_amdgcn_wmma_f32_16x16x32_f16(
                false, a, false, b, (short)0, acc, false, false);
        }
        // tanh epilogue: f32 to out, f16 to next h buffer
#pragma unroll
        for (int r = 0; r < 8; ++r) {
            const int m = mrow + sub * 8 + r;
            const float v = tanhf(acc[r]);
            out[(size_t)m * (T_ * H_) + (size_t)t * H_ + col] = v;
            hout[(size_t)m * H_ + col] = (_Float16)v;
        }

        // device-wide step barrier (monotonic counter, zeroed by prep each call)
        __syncthreads();
        if (tid == 0) {
            __threadfence();                        // release h writes
            atomicAdd(counter, 1u);
            const unsigned target = nwg * (unsigned)(t + 1);
            while (__hip_atomic_load(counter, __ATOMIC_ACQUIRE,
                                     __HIP_MEMORY_SCOPE_AGENT) < target)
                __builtin_amdgcn_s_sleep(1);
        }
        __syncthreads();
        __threadfence();                            // acquire: drop stale h in L0/WGP$
    }
}

// ---------------------------------------------------------------------------
extern "C" void kernel_launch(void* const* d_in, const int* in_sizes, int n_in,
                              void* d_out, int out_size, void* d_ws, size_t ws_size,
                              hipStream_t stream) {
    (void)in_sizes; (void)n_in; (void)out_size;
    if (ws_size < (size_t)WS_NEED) return;

    const float* x  = (const float*)d_in[0];   // [64,512,512]
    const float* h0 = (const float*)d_in[1];   // [64,512]
    const float* U  = (const float*)d_in[2];   // [512,512]
    const float* V  = (const float*)d_in[3];   // [512,512]
    const float* bi = (const float*)d_in[4];   // [512]
    float* out = (float*)d_out;                // [64,512,512]

    char* ws = (char*)d_ws;
    _Float16* Ut = (_Float16*)(ws + WS_UT);
    _Float16* Vt = (_Float16*)(ws + WS_VT);
    _Float16* hb = (_Float16*)(ws + WS_H);
    unsigned* cnt = (unsigned*)(ws + WS_CNT);

    prep_kernel<<<256, 256, 0, stream>>>(U, V, h0, Ut, Vt, hb, cnt);
    gemm_xu_kernel<<<dim3((M_TOTAL / 128) * (H_ / 128)), 256, 0, stream>>>(x, Ut, bi, out);
    rnn_persistent_kernel<<<16, 256, 0, stream>>>(Vt, hb, out, cnt, 16u);
}
